// GraphAttention_CL_67860483277322
// MI455X (gfx1250) — matrix operand order repaired and checked by reading, not compiled, not run on hardware
//
#include <hip/hip_runtime.h>
#include <hip/hip_bf16.h>

typedef __attribute__((ext_vector_type(16))) _Float16 v16h;
typedef __attribute__((ext_vector_type(8)))  _Float16 v8h;
typedef __attribute__((ext_vector_type(8)))  float    v8f;
typedef __attribute__((ext_vector_type(4)))  float    v4f;   // native vector for nontemporal

#define V_    8000
#define D_    256
#define B_    64
#define L_    64
#define VPAD  8192          // padded K so 8 chunks of 1024 cover V (zero pad)
#define CHUNK 1024
#define NCHUNK 8
#define CONC  5.0f

// ---------------- workspace layout (bytes) ----------------
// kbT   : f16 [D_][VPAD]      4,194,304
// rm    : f32 [B_][D_]           65,536
// ctx   : f32 [D_]                1,024
// ctxn  : f32 [1] (padded)          256
// wrel  : f32 [V_]               32,000
// waff  : f32 [V_]               32,000
// part  : f32 [4][B_][D_]       262,144
#define OFF_RM    (4194304)
#define OFF_CTX   (OFF_RM + 65536)
#define OFF_CTXN  (OFF_CTX + 1024)
#define OFF_WREL  (OFF_CTXN + 256)
#define OFF_WAFF  (OFF_WREL + 32000)
#define OFF_PART  (OFF_WAFF + 32000)

// ---- 1) per-batch masked row mean of gathered embeddings ----
__global__ void k_row_mean(const int* __restrict__ clist, const int* __restrict__ clen,
                           const float* __restrict__ embw, float* __restrict__ rm) {
  int b = blockIdx.x, d = threadIdx.x;
  int len = clen[b];
  float s = 0.f;
  for (int l = 0; l < len; ++l) {
    int c = clist[b * L_ + l];
    s += embw[(size_t)c * D_ + d];
  }
  float denom = (float)(len > 0 ? len : 1);
  rm[b * D_ + d] = s / denom;
}

// ---- 2) context vector + its L2 norm ----
__global__ void k_ctx(const float* __restrict__ rm, float* __restrict__ ctx,
                      float* __restrict__ ctxn) {
  int d = threadIdx.x;
  float s = 0.f;
  for (int b = 0; b < B_; ++b) s += rm[b * D_ + d];
  float c = s * (1.f / (float)B_);
  ctx[d] = c;
  __shared__ float red[D_];
  red[d] = c * c;
  __syncthreads();
  for (int off = D_ / 2; off > 0; off >>= 1) {
    if (d < off) red[d] += red[d + off];
    __syncthreads();
  }
  if (d == 0) *ctxn = sqrtf(red[0]);
}

// ---- 3) fold cosine similarity + lambda into per-v coefficients ----
__global__ void k_cos(const float* __restrict__ kb, const float* __restrict__ ctx,
                      const float* __restrict__ ctxn, const float* __restrict__ lam,
                      const float* __restrict__ aff,
                      float* __restrict__ wrel, float* __restrict__ waff) {
  int wave = threadIdx.x >> 5, lane = threadIdx.x & 31;
  int v = blockIdx.x * 8 + wave;
  if (v >= V_) return;
  const float* row = kb + (size_t)v * D_;
  float dot = 0.f, nn = 0.f;
  for (int i = lane; i < D_; i += 32) {
    float x = row[i];
    dot += x * ctx[i];
    nn  += x * x;
  }
  for (int off = 16; off > 0; off >>= 1) {
    dot += __shfl_down(dot, off, 32);
    nn  += __shfl_down(nn,  off, 32);
  }
  if (lane == 0) {
    float cd   = fmaxf(sqrtf(nn) * (*ctxn), 1e-8f);
    float cosv = fabsf(dot) / cd;
    float lv   = lam[v];
    wrel[v] = lv * cosv;
    waff[v] = (1.f - lv) * aff[v];
  }
}

// ---- 4) kbT[n][v] = f16(kb[v][n]), zero-padded to VPAD ----
__global__ void k_kbT(const float* __restrict__ kb, _Float16* __restrict__ kbT) {
  size_t idx   = (size_t)blockIdx.x * blockDim.x + threadIdx.x;
  size_t total = (size_t)D_ * VPAD;
  size_t step  = (size_t)gridDim.x * blockDim.x;
  for (; idx < total; idx += step) {
    int n = (int)(idx >> 13);            // VPAD == 8192
    int v = (int)(idx & (VPAD - 1));
    kbT[idx] = (v < V_) ? (_Float16)kb[(size_t)v * D_ + n] : (_Float16)0.f;
  }
}

// ---- 5) broadcast embed_kb_w -> local_nodes [B][V][D], non-temporal ----
__global__ void k_local_nodes(const float* __restrict__ kb, float* __restrict__ outln) {
  const int nvec = V_ * D_ / 4;          // v4f count per copy
  const v4f* src = (const v4f*)kb;
  v4f* dst = (v4f*)outln;
  int step = gridDim.x * blockDim.x;
  for (int i = blockIdx.x * blockDim.x + threadIdx.x; i < nvec; i += step) {
    v4f x = src[i];
    for (int b = 0; b < B_; ++b)
      __builtin_nontemporal_store(x, &dst[(size_t)b * nvec + i]);
  }
}

// weight -> pre-softmax logit z = CONC * w
__device__ __forceinline__ float logit_z(float e, const int* __restrict__ maskb,
                                         const float* __restrict__ wrel,
                                         const float* __restrict__ waff, int v) {
  float w = e * wrel[v] + ((e > 0.f) ? waff[v] : 0.f);
  if (maskb[v] == 0) w = 1e-9f;
  return CONC * w;
}

// ---- 6) softmax over V + attn @ kbT via v_wmma_f32_16x16x32_f16 ----
// one block per (b, 16-row M tile); wave w owns output columns [32w, 32w+32)
__global__ void k_attn_gemm(const int* __restrict__ clist, const int* __restrict__ clen,
                            const int* __restrict__ cmask, const float* __restrict__ edge,
                            const float* __restrict__ wrel, const float* __restrict__ waff,
                            const _Float16* __restrict__ kbT, float* __restrict__ partial) {
  __shared__ _Float16 p_tile[16][CHUNK];   // 32 KB attention-prob tile (f16)
  __shared__ float s_max[16], s_sum[16];
  __shared__ int   s_crow[16];

  const int b     = blockIdx.x >> 2;
  const int mtile = blockIdx.x & 3;
  const int tid   = threadIdx.x;
  const int wave  = tid >> 5, lane = tid & 31;
  const int len   = clen[b];
  float* pout = partial + ((size_t)mtile * B_ + b) * D_;

  if (mtile * 16 >= len) { pout[tid] = 0.f; return; }   // block-uniform skip
  const float inv_denom = 1.f / (float)len;             // len >= 1 here
  const int* maskb = cmask + (size_t)b * V_;

  if (tid < 16) s_crow[tid] = clist[b * L_ + mtile * 16 + tid];
  __syncthreads();

  // ---- softmax stats: each wave handles 2 rows, two streaming passes over V
  for (int rr = 0; rr < 2; ++rr) {
    int row = wave * 2 + rr;
    const float* erow = edge + (size_t)s_crow[row] * V_;
    float zmax = -3.4e38f;
    for (int v = lane; v < V_; v += 32) {
      __builtin_prefetch(erow + v + 512, 0, 1);
      zmax = fmaxf(zmax, logit_z(erow[v], maskb, wrel, waff, v));
    }
    for (int off = 16; off > 0; off >>= 1)
      zmax = fmaxf(zmax, __shfl_xor(zmax, off, 32));
    float zsum = 0.f;
    for (int v = lane; v < V_; v += 32)
      zsum += __expf(logit_z(erow[v], maskb, wrel, waff, v) - zmax);
    for (int off = 16; off > 0; off >>= 1)
      zsum += __shfl_xor(zsum, off, 32);
    if (lane == 0) { s_max[row] = zmax; s_sum[row] = zsum; }
  }
  __syncthreads();

  v8f acc0 = {0.f, 0.f, 0.f, 0.f, 0.f, 0.f, 0.f, 0.f};
  v8f acc1 = {0.f, 0.f, 0.f, 0.f, 0.f, 0.f, 0.f, 0.f};

  const int half8 = (lane < 16) ? 0 : 8;    // ISA 16-bit A / f32 C half-wave split
  const int arow  = lane & 15;
  const int ncol0 = wave * 32 + (lane & 15);
  const _Float16* bptr0 = kbT + (size_t)ncol0 * VPAD;          // N-tile 0 column
  const _Float16* bptr1 = kbT + (size_t)(ncol0 + 16) * VPAD;   // N-tile 1 column

  for (int chunk = 0; chunk < NCHUNK; ++chunk) {
    const int vbase = chunk * CHUNK;
    // fill probability tile (f16), zero-pad past V_
    for (int idx = tid; idx < 16 * CHUNK; idx += 256) {
      int row = idx >> 10;                 // CHUNK == 1024
      int col = idx & (CHUNK - 1);
      int v = vbase + col;
      float p = 0.f;
      if (v < V_) {
        float z = logit_z(edge[(size_t)s_crow[row] * V_ + v], maskb, wrel, waff, v);
        p = __expf(z - s_max[row]) / s_sum[row];
      }
      p_tile[row][col] = (_Float16)p;
    }
    __syncthreads();

    for (int ks = 0; ks < CHUNK / 32; ++ks) {
      const int k0 = ks * 32;
      // A fragment: lanes 0-15 K{0..15}, lanes 16-31 K{16..31} (matches B)
      v8h alo = *(const v8h*)&p_tile[arow][k0 + 2 * half8];
      v8h ahi = *(const v8h*)&p_tile[arow][k0 + 2 * half8 + 8];
      v16h a = __builtin_shufflevector(alo, ahi,
                 0, 1, 2, 3, 4, 5, 6, 7, 8, 9, 10, 11, 12, 13, 14, 15);
      // B fragment: lane -> column n, 16 consecutive K values from kbT
      const int kb16 = vbase + k0 + ((lane < 16) ? 0 : 16);
      v16h b0 = *(const v16h*)(bptr0 + kb16);
      v16h b1 = *(const v16h*)(bptr1 + kb16);
      acc0 = __builtin_amdgcn_wmma_f32_16x16x32_f16(false, a, false, b0,
                                                    (short)0, acc0, false, false);
      acc1 = __builtin_amdgcn_wmma_f32_16x16x32_f16(false, a, false, b1,
                                                    (short)0, acc1, false, false);
    }
    __syncthreads();
  }

  // ---- weighted row reduction of C tiles (mask invalid l, divide by len)
  float s0 = 0.f, s1 = 0.f;
#pragma unroll
  for (int r = 0; r < 8; ++r) {
    int m = mtile * 16 + r + half8;       // C layout: lanes>=16 hold M = r+8
    float wrow = (m < len) ? inv_denom : 0.f;
    s0 += wrow * acc0[r];
    s1 += wrow * acc1[r];
  }
  s0 += __shfl_xor(s0, 16, 32);
  s1 += __shfl_xor(s1, 16, 32);
  if (lane < 16) {
    pout[wave * 32 + lane]      = s0;     // distinct columns per wave: race-free
    pout[wave * 32 + 16 + lane] = s1;
  }
}

// ---- 7) deterministic sum of the 4 M-tile partials ----
__global__ void k_finalize(const float* __restrict__ partial, float* __restrict__ out) {
  int i = blockIdx.x * D_ + threadIdx.x;
  float s = 0.f;
  for (int t = 0; t < 4; ++t) s += partial[(size_t)t * B_ * D_ + i];
  out[i] = s;
}

extern "C" void kernel_launch(void* const* d_in, const int* in_sizes, int n_in,
                              void* d_out, int out_size, void* d_ws, size_t ws_size,
                              hipStream_t stream) {
  const int*   clist = (const int*)  d_in[0];
  const int*   clen  = (const int*)  d_in[1];
  const int*   cmask = (const int*)  d_in[2];
  const float* edge  = (const float*)d_in[3];
  const float* aff   = (const float*)d_in[4];
  const float* embw  = (const float*)d_in[5];
  const float* kb    = (const float*)d_in[6];
  const float* lam   = (const float*)d_in[7];
  float* out = (float*)d_out;                 // [B,D] then [B,V,D], all f32

  char* ws = (char*)d_ws;                     // needs ~4.6 MB
  _Float16* kbT  = (_Float16*)ws;
  float* rm      = (float*)(ws + OFF_RM);
  float* ctx     = (float*)(ws + OFF_CTX);
  float* ctxn    = (float*)(ws + OFF_CTXN);
  float* wrel    = (float*)(ws + OFF_WREL);
  float* waff    = (float*)(ws + OFF_WAFF);
  float* partial = (float*)(ws + OFF_PART);

  k_row_mean   <<<B_,      D_,  0, stream>>>(clist, clen, embw, rm);
  k_ctx        <<<1,       D_,  0, stream>>>(rm, ctx, ctxn);
  k_cos        <<<V_ / 8,  256, 0, stream>>>(kb, ctx, ctxn, lam, aff, wrel, waff);
  k_kbT        <<<2048,    256, 0, stream>>>(kb, kbT);
  k_attn_gemm  <<<B_ * 4,  256, 0, stream>>>(clist, clen, cmask, edge, wrel, waff,
                                             kbT, partial);
  k_finalize   <<<B_,      D_,  0, stream>>>(partial, out);
  k_local_nodes<<<2048,    256, 0, stream>>>(kb, out + B_ * D_);
}
